// OverlapAttention_28217935134733
// MI455X (gfx1250) — matrix operand order repaired and checked by reading, not compiled, not run on hardware
//
#include <hip/hip_runtime.h>

// ---------------------------------------------------------------------------
// Problem geometry (matches reference)
// ---------------------------------------------------------------------------
#define SEQ        2048
#define BATCH      2
#define HIDDEN     2048
#define NUM_HEADS  16
#define NUM_KV     4
#define HEAD_DIM   128
#define QPG        4                       // Q heads per KV head
#define GROUP_SZ   ((QPG + 2) * HEAD_DIM)  // 768
#define TOTAL_PROJ (NUM_KV * GROUP_SZ)     // 3072
#define MROWS      (SEQ * BATCH)           // 4096

// ---------------------------------------------------------------------------
// CDNA5 WMMA types / helpers (wave32, 16x16x32 bf16 -> f32)
// ---------------------------------------------------------------------------
typedef unsigned int u32x4   __attribute__((ext_vector_type(4)));
typedef __bf16       bf16x16 __attribute__((ext_vector_type(16)));
typedef float        f32x8   __attribute__((ext_vector_type(8)));

union BFrag { bf16x16 v; u32x4 q[2]; };

__device__ __forceinline__ unsigned short f2bf(float f) {
  unsigned int u = __float_as_uint(f);
  u += 0x7FFFu + ((u >> 16) & 1u);      // round-to-nearest-even
  return (unsigned short)(u >> 16);
}

__device__ __forceinline__ f32x8 wmma_bf16(bf16x16 a, bf16x16 b, f32x8 c) {
  return __builtin_amdgcn_wmma_f32_16x16x32_bf16(
      /*neg_a=*/false, a, /*neg_b=*/false, b,
      /*c_mod=*/(short)0, c, /*reuse_a=*/false, /*reuse_b=*/false);
}

// CDNA5 async memory->LDS DMA (ASYNCcnt-tracked), via inline asm
__device__ __forceinline__ void async_ld_b128(unsigned lds_off, const void* gaddr) {
  asm volatile("global_load_async_to_lds_b128 %0, %1, off"
               :: "v"(lds_off), "v"(gaddr) : "memory");
}

// ---------------------------------------------------------------------------
// Workspace layout (ushort/bf16 element offsets)
// ---------------------------------------------------------------------------
constexpr size_t OFF_XB = 0;                                        // [4096][2048]
constexpr size_t OFF_WQ = OFF_XB + (size_t)MROWS * HIDDEN;          // [3072][2048]
constexpr size_t OFF_WP = OFF_WQ + (size_t)TOTAL_PROJ * HIDDEN;     // [2048][2048]
constexpr size_t OFF_Q  = OFF_WP + (size_t)HIDDEN * HIDDEN;         // [2][16][2048][128]
constexpr size_t OFF_K  = OFF_Q  + (size_t)BATCH * NUM_HEADS * SEQ * HEAD_DIM;
constexpr size_t OFF_VT = OFF_K  + (size_t)BATCH * NUM_KV * SEQ * HEAD_DIM;
constexpr size_t OFF_AT = OFF_VT + (size_t)BATCH * NUM_KV * HEAD_DIM * SEQ;

// ---------------------------------------------------------------------------
// Kernel 1: fp32 -> bf16 conversion of x, w_qkv, w_proj
// ---------------------------------------------------------------------------
__global__ void cvt_kernel(const float* __restrict__ x,
                           const float* __restrict__ wq,
                           const float* __restrict__ wp,
                           unsigned short* __restrict__ xb,
                           unsigned short* __restrict__ wqb,
                           unsigned short* __restrict__ wpb) {
  const size_t tid    = (size_t)blockIdx.x * blockDim.x + threadIdx.x;
  const size_t stride = (size_t)gridDim.x * blockDim.x;
  for (size_t i = tid; i < (size_t)MROWS * HIDDEN; i += stride)      xb[i]  = f2bf(x[i]);
  for (size_t i = tid; i < (size_t)TOTAL_PROJ * HIDDEN; i += stride) wqb[i] = f2bf(wq[i]);
  for (size_t i = tid; i < (size_t)HIDDEN * HIDDEN; i += stride)     wpb[i] = f2bf(wp[i]);
}

// ---------------------------------------------------------------------------
// 32x64 register-tiled WMMA GEMM core (K contiguous for both operands)
//   C[m,n] = sum_k A[m,k] * B[n,k];  2 A-frags x 4 B-frags -> 8 accumulators
// ---------------------------------------------------------------------------
__device__ __forceinline__ void wave_gemm32x64(const unsigned short* __restrict__ A,
                                               const unsigned short* __restrict__ B,
                                               int mTile, int nTile, int Kdim,
                                               int l15, int hl, f32x8 c[2][4]) {
  const unsigned short* ap0 = A + (size_t)(mTile + l15) * Kdim;
  const unsigned short* ap1 = ap0 + (size_t)16 * Kdim;
  const unsigned short* bp[4];
#pragma unroll
  for (int ni = 0; ni < 4; ++ni) bp[ni] = B + (size_t)(nTile + ni * 16 + l15) * Kdim;
  const int akb = hl * 8;    // A frag: runs [akb..akb+7], [akb+16..akb+23]
  const int bkb = hl * 8;    // B frag: runs [bkb..bkb+7], [bkb+16..bkb+23]

  const f32x8 z = {};
#pragma unroll
  for (int mi = 0; mi < 2; ++mi)
#pragma unroll
    for (int ni = 0; ni < 4; ++ni) c[mi][ni] = z;

  for (int k = 0; k < Kdim; k += 32) {
    __builtin_prefetch(ap0 + k + 512, 0, 3);     // global_prefetch (WGP temporal)
    __builtin_prefetch(bp[0] + k + 512, 0, 3);
    BFrag a[2], bb[4];
    a[0].q[0] = *(const u32x4*)(ap0 + k + akb);
    a[0].q[1] = *(const u32x4*)(ap0 + k + akb + 16);
    a[1].q[0] = *(const u32x4*)(ap1 + k + akb);
    a[1].q[1] = *(const u32x4*)(ap1 + k + akb + 16);
#pragma unroll
    for (int ni = 0; ni < 4; ++ni) {
      bb[ni].q[0] = *(const u32x4*)(bp[ni] + k + bkb);
      bb[ni].q[1] = *(const u32x4*)(bp[ni] + k + bkb + 16);
    }
#pragma unroll
    for (int mi = 0; mi < 2; ++mi)
#pragma unroll
      for (int ni = 0; ni < 4; ++ni)
        c[mi][ni] = wmma_bf16(a[mi].v, bb[ni].v, c[mi][ni]);
  }
}

// ---------------------------------------------------------------------------
// Kernel 2: QKV projection GEMM + scatter into Q / K / V^T (bf16)
// ---------------------------------------------------------------------------
__device__ __forceinline__ void scatter_qkv(const f32x8& c, int mBase, int pBase,
                                            int l15, int hl,
                                            unsigned short* __restrict__ Q,
                                            unsigned short* __restrict__ K,
                                            unsigned short* __restrict__ Vt) {
  const int p = pBase + l15;
  const int g = p / GROUP_SZ;
  const int o = p - g * GROUP_SZ;
  const float qscale = 0.08838834764831845f;  // 1/sqrt(128)
#pragma unroll
  for (int r = 0; r < 8; ++r) {
    const int m = mBase + r + hl * 8;
    const int s = m >> 1, b = m & 1;
    const float val = c[r];
    if (o < QPG * HEAD_DIM) {
      const int h = g * QPG + (o >> 7), d = o & (HEAD_DIM - 1);
      Q[((size_t)(b * NUM_HEADS + h) * SEQ + s) * HEAD_DIM + d] = f2bf(val * qscale);
    } else if (o < (QPG + 1) * HEAD_DIM) {
      const int d = o - QPG * HEAD_DIM;
      K[((size_t)(b * NUM_KV + g) * SEQ + s) * HEAD_DIM + d] = f2bf(val);
    } else {
      const int d = o - (QPG + 1) * HEAD_DIM;
      Vt[((size_t)(b * NUM_KV + g) * HEAD_DIM + d) * SEQ + s] = f2bf(val);
    }
  }
}

__global__ __launch_bounds__(128) void qkv_gemm_kernel(
    const unsigned short* __restrict__ Xb, const unsigned short* __restrict__ Wq,
    unsigned short* __restrict__ Q, unsigned short* __restrict__ K,
    unsigned short* __restrict__ Vt) {
  const int lane = threadIdx.x & 31, wave = threadIdx.x >> 5;
  const int l15 = lane & 15, hl = lane >> 4;
  const int mTile = blockIdx.y * 32;
  const int nTile = blockIdx.x * 256 + wave * 64;

  f32x8 c[2][4];
  wave_gemm32x64(Xb, Wq, mTile, nTile, HIDDEN, l15, hl, c);

#pragma unroll
  for (int mi = 0; mi < 2; ++mi)
#pragma unroll
    for (int ni = 0; ni < 4; ++ni)
      scatter_qkv(c[mi][ni], mTile + mi * 16, nTile + ni * 16, l15, hl, Q, K, Vt);
}

// ---------------------------------------------------------------------------
// Kernel 3: flash attention with double-buffered async K/V staging in LDS.
//   4 waves of a block share one (b,h) head; K/V tiles for step t+32 are
//   DMA'd (global_load_async_to_lds_b128 / ASYNCcnt) while computing step t.
// LDS row padding: K rows 272B (68 dwords), V rows 80B (20 dwords) -> the 16
// fragment-reading lanes land on distinct banks.
// ---------------------------------------------------------------------------
#define KROW 136   // ushorts per padded K row (128 + 8)
#define VROW 40    // ushorts per padded V row (32 + 8)

__device__ __forceinline__ void attn_stage_async(
    const unsigned short* __restrict__ kTile,   // K rows t..t+31 (contiguous 8KB)
    const unsigned short* __restrict__ vbase,   // V^T head base
    int t, unsigned ldsKb, unsigned ldsVb, int tid) {
#pragma unroll
  for (int j = 0; j < 4; ++j) {
    const int idx = tid + j * 128;                       // 0..511, 16B chunks
    {  // K: 16 chunks per 256B row
      const int row = idx >> 4, within = (idx & 15) << 4;
      async_ld_b128(ldsKb + (unsigned)(row * (KROW * 2) + within),
                    (const char*)kTile + idx * 16);
    }
    {  // V: 4 chunks per 64B row
      const int d = idx >> 2, within = (idx & 3) << 4;
      async_ld_b128(ldsVb + (unsigned)(d * (VROW * 2) + within),
                    (const char*)(vbase + (size_t)d * SEQ + t) + within);
    }
  }
}

__global__ __launch_bounds__(128) void attn_kernel(
    const unsigned short* __restrict__ Q, const unsigned short* __restrict__ Kt,
    const unsigned short* __restrict__ Vt, unsigned short* __restrict__ O) {
  __shared__ __align__(16) unsigned short ldsK[2][32 * KROW];   // 2 x 8.5KB
  __shared__ __align__(16) unsigned short ldsV[2][128 * VROW];  // 2 x 10KB
  __shared__ __align__(16) unsigned short ldsP[4][16][32];      // per-wave P tile

  const int tid = threadIdx.x;
  const int lane = tid & 31, wave = tid >> 5;
  const int l15 = lane & 15, hl = lane >> 4;
  const int bh = blockIdx.y;              // 0..31
  const int b = bh >> 4, h = bh & 15, hk = h >> 2;
  const int sTile = (blockIdx.x * 4 + wave) * 16;

  const unsigned short* qrow  = Q  + ((size_t)(b * NUM_HEADS + h) * SEQ + sTile + l15) * HEAD_DIM;
  const unsigned short* kbase = Kt + (size_t)(b * NUM_KV + hk) * SEQ * HEAD_DIM;
  const unsigned short* vbase = Vt + (size_t)(b * NUM_KV + hk) * HEAD_DIM * SEQ;

  const unsigned ldsKb[2] = {(unsigned)(uintptr_t)&ldsK[0][0], (unsigned)(uintptr_t)&ldsK[1][0]};
  const unsigned ldsVb[2] = {(unsigned)(uintptr_t)&ldsV[0][0], (unsigned)(uintptr_t)&ldsV[1][0]};

  // Q tile: 4 A-fragments covering d = 0..127 (k=32 each)
  BFrag qf[4];
#pragma unroll
  for (int dk = 0; dk < 4; ++dk) {
    const int off = dk * 32 + hl * 8;
    qf[dk].q[0] = *(const u32x4*)(qrow + off);
    qf[dk].q[1] = *(const u32x4*)(qrow + off + 16);
  }

  const f32x8 z = {};
  f32x8 o_acc[8];
#pragma unroll
  for (int n = 0; n < 8; ++n) o_acc[n] = z;
  float mrow[8], lrow[8];
#pragma unroll
  for (int r = 0; r < 8; ++r) { mrow[r] = -3.0e38f; lrow[r] = 0.f; }

  // prologue: stage tile 0 into buffer 0
  attn_stage_async(kbase, vbase, 0, ldsKb[0], ldsVb[0], tid);

  const int NT = SEQ / 32;                // 64 t-steps
  for (int ti = 0; ti < NT; ++ti) {
    const int t = ti * 32;
    const int cur = ti & 1;

    if (ti + 1 < NT) {  // DMA next tile into the other buffer (overlapped)
      attn_stage_async(kbase + (size_t)(t + 32) * HEAD_DIM, vbase, t + 32,
                       ldsKb[cur ^ 1], ldsVb[cur ^ 1], tid);
      asm volatile("s_wait_asynccnt 0x8" ::: "memory");  // oldest 8 done (in-order)
    } else {
      asm volatile("s_wait_asynccnt 0x0" ::: "memory");
    }
    __syncthreads();                       // current tile visible to all waves

    const unsigned short* lk = &ldsK[cur][0];
    const unsigned short* lv = &ldsV[cur][0];

    // ---- scores: two 16x16 tiles from LDS K ----
    f32x8 s0 = z, s1 = z;
#pragma unroll
    for (int dk = 0; dk < 4; ++dk) {
      const int kb = hl * 8 + dk * 32;     // ushort offset within K row
      const unsigned short* kr0 = lk + l15 * KROW + kb;
      const unsigned short* kr1 = lk + (16 + l15) * KROW + kb;
      BFrag k0, k1;
      k0.q[0] = *(const u32x4*)(kr0);
      k0.q[1] = *(const u32x4*)(kr0 + 16);
      k1.q[0] = *(const u32x4*)(kr1);
      k1.q[1] = *(const u32x4*)(kr1 + 16);
      s0 = wmma_bf16(qf[dk].v, k0.v, s0);
      s1 = wmma_bf16(qf[dk].v, k1.v, s1);
    }

    // ---- online softmax (row r lives in one 16-lane half) ----
    float corr[8];
#pragma unroll
    for (int r = 0; r < 8; ++r) {
      float v = fmaxf(s0[r], s1[r]);
#pragma unroll
      for (int off = 1; off <= 8; off <<= 1) v = fmaxf(v, __shfl_xor(v, off, 32));
      const float mn = fmaxf(mrow[r], v);
      corr[r] = __expf(mrow[r] - mn);
      mrow[r] = mn;
      const float p0 = __expf(s0[r] - mn);
      const float p1 = __expf(s1[r] - mn);
      s0[r] = p0; s1[r] = p1;
      float sum = p0 + p1;
#pragma unroll
      for (int off = 1; off <= 8; off <<= 1) sum += __shfl_xor(sum, off, 32);
      lrow[r] = lrow[r] * corr[r] + sum;
    }
#pragma unroll
    for (int n = 0; n < 8; ++n)
#pragma unroll
      for (int r = 0; r < 8; ++r) o_acc[n][r] *= corr[r];

    // ---- re-layout P: C-fragment -> A-fragment via LDS ----
#pragma unroll
    for (int r = 0; r < 8; ++r) {
      ldsP[wave][r + hl * 8][l15]      = f2bf(s0[r]);
      ldsP[wave][r + hl * 8][l15 + 16] = f2bf(s1[r]);
    }
    __syncthreads();
    BFrag pf;
    pf.q[0] = *(const u32x4*)&ldsP[wave][l15][hl * 8];
    pf.q[1] = *(const u32x4*)&ldsP[wave][l15][hl * 8 + 16];

    // ---- O += P * V  from LDS V ----
#pragma unroll
    for (int n = 0; n < 8; ++n) {
      const unsigned short* vr = lv + (n * 16 + l15) * VROW + hl * 8;
      BFrag vf;
      vf.q[0] = *(const u32x4*)(vr);
      vf.q[1] = *(const u32x4*)(vr + 16);
      o_acc[n] = wmma_bf16(pf.v, vf.v, o_acc[n]);
    }
    __syncthreads();   // everyone done with buf[cur] + ldsP before next overwrite
  }

  // ---- epilogue: normalize, store bf16 at [m = s*BATCH+b][e = h*128+d] ----
#pragma unroll
  for (int r = 0; r < 8; ++r) {
    const float inv = 1.0f / lrow[r];
    const int s = sTile + r + hl * 8;
    const size_t mrowi = (size_t)(s * BATCH + b) * (NUM_HEADS * HEAD_DIM);
#pragma unroll
    for (int n = 0; n < 8; ++n)
      O[mrowi + h * HEAD_DIM + n * 16 + l15] = f2bf(o_acc[n][r] * inv);
  }
}

// ---------------------------------------------------------------------------
// Kernel 4: output projection GEMM, f32 result straight into d_out
// ---------------------------------------------------------------------------
__global__ __launch_bounds__(128) void proj_gemm_kernel(
    const unsigned short* __restrict__ A, const unsigned short* __restrict__ Wp,
    float* __restrict__ out) {
  const int lane = threadIdx.x & 31, wave = threadIdx.x >> 5;
  const int l15 = lane & 15, hl = lane >> 4;
  const int mTile = blockIdx.y * 32;
  const int nTile = blockIdx.x * 256 + wave * 64;

  f32x8 c[2][4];
  wave_gemm32x64(A, Wp, mTile, nTile, HIDDEN, l15, hl, c);

#pragma unroll
  for (int mi = 0; mi < 2; ++mi)
#pragma unroll
    for (int r = 0; r < 8; ++r) {
      const size_t row = (size_t)(mTile + mi * 16 + r + hl * 8) * HIDDEN;
#pragma unroll
      for (int ni = 0; ni < 4; ++ni)
        out[row + nTile + ni * 16 + l15] = c[mi][ni][r];
    }
}

// ---------------------------------------------------------------------------
// Host launcher
// ---------------------------------------------------------------------------
extern "C" void kernel_launch(void* const* d_in, const int* in_sizes, int n_in,
                              void* d_out, int out_size, void* d_ws, size_t ws_size,
                              hipStream_t stream) {
  const float* x  = (const float*)d_in[0];   // [2048][2][2048]
  const float* wq = (const float*)d_in[1];   // [3072][2048]
  const float* wp = (const float*)d_in[2];   // [2048][2048]
  float* out = (float*)d_out;                // [2048][2][2048]

  unsigned short* ws = (unsigned short*)d_ws;
  unsigned short* Xb  = ws + OFF_XB;
  unsigned short* Wqb = ws + OFF_WQ;
  unsigned short* Wpb = ws + OFF_WP;
  unsigned short* Qb  = ws + OFF_Q;
  unsigned short* Kb  = ws + OFF_K;
  unsigned short* Vtb = ws + OFF_VT;
  unsigned short* At  = ws + OFF_AT;

  // 1) fp32 -> bf16
  cvt_kernel<<<2048, 256, 0, stream>>>(x, wq, wp, Xb, Wqb, Wpb);

  // 2) QKV GEMM (M=4096, N=3072, K=2048), scatter to Q/K/V^T
  qkv_gemm_kernel<<<dim3(TOTAL_PROJ / 256, MROWS / 32), 128, 0, stream>>>(
      Xb, Wqb, Qb, Kb, Vtb);

  // 3) flash attention: 32 q-tile groups x 32 (b,h) heads
  attn_kernel<<<dim3(SEQ / 64, BATCH * NUM_HEADS), 128, 0, stream>>>(
      Qb, Kb, Vtb, At);

  // 4) output projection (M=4096, N=2048, K=2048) -> f32
  proj_gemm_kernel<<<dim3(HIDDEN / 256, MROWS / 32), 128, 0, stream>>>(
      At, Wpb, out);
}
